// Attention_79448305041876
// MI455X (gfx1250) — matrix operand order repaired and checked by reading, not compiled, not run on hardware
//
#include <hip/hip_runtime.h>

typedef __bf16 bf16_t;
typedef __bf16 v16bf __attribute__((ext_vector_type(16)));
typedef __bf16 v8bf  __attribute__((ext_vector_type(8)));
typedef float  v8f   __attribute__((ext_vector_type(8)));

#define B_   8
#define S_   1024
#define H_   16
#define KV_  4
#define D_   128
#define G_   (H_ / KV_)
#define QB   128
#define NQB  (S_ / QB)
#define SCALE_F 0.08838834764831845f
#define LOG2E_F 1.4426950408889634f

// ---- LDS fragment loaders (16-bit A/B layouts per CDNA5 ISA 7.12.2) ----
// A 16x32 (bf16): lane half 0 -> K = c*32 + [0..7] and [16..23]; half 1 -> +8
__device__ __forceinline__ v16bf ld_a(const bf16_t* row, int c, int half) {
  const bf16_t* p = row + c * 32 + half * 8;
  v8bf lo = *(const v8bf*)(p);
  v8bf hi = *(const v8bf*)(p + 16);
  return __builtin_shufflevector(lo, hi, 0,1,2,3,4,5,6,7,8,9,10,11,12,13,14,15);
}
// B 32x16 (bf16): lane half 0 -> K = c*32 + [0..7] and [16..23]; half 1 -> +8
__device__ __forceinline__ v16bf ld_b(const bf16_t* row, int c, int half) {
  const bf16_t* p = row + c * 32 + half * 8;
  v8bf lo = *(const v8bf*)(p);
  v8bf hi = *(const v8bf*)(p + 16);
  return __builtin_shufflevector(lo, hi, 0,1,2,3,4,5,6,7,8,9,10,11,12,13,14,15);
}

// ---------------- cache zero + scatter ----------------
__global__ void zero_kernel(float4* __restrict__ p, int n4) {
  int i = blockIdx.x * blockDim.x + threadIdx.x;
  if (i < n4) p[i] = make_float4(0.f, 0.f, 0.f, 0.f);
}

__global__ void scatter_kernel(const float4* __restrict__ k4,
                               const float4* __restrict__ v4,
                               const int* __restrict__ slots,
                               float4* __restrict__ kc4,
                               float4* __restrict__ vc4) {
  int i = blockIdx.x * blockDim.x + threadIdx.x;  // T * (KV*D/4) = 8192*128
  if (i >= (B_ * S_) * (KV_ * D_ / 4)) return;
  int t   = i >> 7;        // 128 float4 per token row
  int off = i & 127;
  int slot = slots[t];
  kc4[(size_t)slot * 128 + off] = k4[i];
  vc4[(size_t)slot * 128 + off] = v4[i];
}

// ---------------- staircase flash attention ----------------
__global__ __launch_bounds__(256)
void attn_kernel(const float* __restrict__ q, const float* __restrict__ k,
                 const float* __restrict__ v, float* __restrict__ out) {
  extern __shared__ bf16_t smem[];
  bf16_t* Qs  = smem;                 // [128][128] q rows (pre-scaled)
  bf16_t* Ks  = smem + 128 * 128;     // [key][d]; aliased with P after barrier
  bf16_t* Vts = smem + 2 * 128 * 128; // [d][key] (transposed V)

  const int wg  = blockIdx.x;
  const int qi  = wg % NQB;
  const int h   = (wg / NQB) % H_;
  const int b   = wg / (NQB * H_);
  const int kvh = h / G_;

  const int tid  = threadIdx.x;
  const int wave = tid >> 5;
  const int lane = tid & 31;
  const int half = lane >> 4;
  const int lm   = lane & 15;

  const float  qscale = SCALE_F * LOG2E_F;  // fold softmax scale + log2(e)
  const size_t qrow0  = (size_t)(b * S_ + qi * QB);

  // ---- stage Q once (f32 -> bf16, pre-scaled) ----
  {
    const float* qb = q + qrow0 * (H_ * D_) + h * D_;
    for (int idx = tid; idx < QB * (D_ / 4); idx += 256) {
      int row = idx >> 5;
      int c4  = (idx & 31) * 4;
      const float4 f = *(const float4*)(qb + (size_t)row * (H_ * D_) + c4);
      bf16_t* dst = Qs + row * D_ + c4;
      dst[0] = (bf16_t)(f.x * qscale);
      dst[1] = (bf16_t)(f.y * qscale);
      dst[2] = (bf16_t)(f.z * qscale);
      dst[3] = (bf16_t)(f.w * qscale);
    }
  }

  v8f o_acc[8];
  #pragma unroll
  for (int n = 0; n < 8; ++n) o_acc[n] = (v8f){0.f,0.f,0.f,0.f,0.f,0.f,0.f,0.f};
  float m_r[8], l_r[8];
  #pragma unroll
  for (int r = 0; r < 8; ++r) { m_r[r] = -3.0e38f; l_r[r] = 0.0f; }

  const bf16_t* qrow_lds = Qs + (wave * 16 + lm) * D_;

  for (int j = 0; j <= qi; ++j) {
    __syncthreads();  // previous iter's P/V reads complete before restage
    // ---- stage K (row-major) and V (transposed) for this key block ----
    {
      const float* kb = k + ((size_t)(b * S_ + j * QB)) * (KV_ * D_) + kvh * D_;
      const float* vb = v + ((size_t)(b * S_ + j * QB)) * (KV_ * D_) + kvh * D_;
      for (int idx = tid; idx < QB * (D_ / 4); idx += 256) {
        int row = idx >> 5;
        int c4  = (idx & 31) * 4;
        const float4 kf = *(const float4*)(kb + (size_t)row * (KV_ * D_) + c4);
        bf16_t* kd = Ks + row * D_ + c4;
        kd[0] = (bf16_t)kf.x; kd[1] = (bf16_t)kf.y;
        kd[2] = (bf16_t)kf.z; kd[3] = (bf16_t)kf.w;
        const float4 vf = *(const float4*)(vb + (size_t)row * (KV_ * D_) + c4);
        Vts[(c4 + 0) * QB + row] = (bf16_t)vf.x;
        Vts[(c4 + 1) * QB + row] = (bf16_t)vf.y;
        Vts[(c4 + 2) * QB + row] = (bf16_t)vf.z;
        Vts[(c4 + 3) * QB + row] = (bf16_t)vf.w;
      }
    }
    __syncthreads();

    // ---- S = Q * K^T  (16x128 per wave, 8 tiles x 4 k-chunks) ----
    v8f s[8];
    #pragma unroll
    for (int t = 0; t < 8; ++t) {
      v8f acc = (v8f){0.f,0.f,0.f,0.f,0.f,0.f,0.f,0.f};
      const bf16_t* krow = Ks + (t * 16 + lm) * D_;
      #pragma unroll
      for (int c = 0; c < 4; ++c) {
        v16bf a  = ld_a(qrow_lds, c, half);
        v16bf bb = ld_b(krow, c, half);
        acc = __builtin_amdgcn_wmma_f32_16x16x32_bf16(
            false, a, false, bb, (short)0, acc, false, false);
      }
      s[t] = acc;
    }

    // ---- online softmax (base 2); row = wave*16 + half*8 + r ----
    float mx[8];
    #pragma unroll
    for (int r = 0; r < 8; ++r) {
      float m0 = s[0][r];
      #pragma unroll
      for (int t = 1; t < 8; ++t) m0 = fmaxf(m0, s[t][r]);
      #pragma unroll
      for (int d = 1; d < 16; d <<= 1) m0 = fmaxf(m0, __shfl_xor(m0, d, 32));
      mx[r] = m0;
    }
    #pragma unroll
    for (int r = 0; r < 8; ++r) {
      float mnew = fmaxf(m_r[r], mx[r]);
      float corr = __builtin_amdgcn_exp2f(m_r[r] - mnew);
      float sum = 0.0f;
      #pragma unroll
      for (int t = 0; t < 8; ++t) {
        float p = __builtin_amdgcn_exp2f(s[t][r] - mnew);
        s[t][r] = p;
        sum += p;
      }
      #pragma unroll
      for (int d = 1; d < 16; d <<= 1) sum += __shfl_xor(sum, d, 32);
      l_r[r] = l_r[r] * corr + sum;
      m_r[r] = mnew;
      #pragma unroll
      for (int n = 0; n < 8; ++n) o_acc[n][r] *= corr;
    }

    __syncthreads();  // all waves done reading Ks -> safe to alias with P
    bf16_t* Ps = Ks;  // [qrow][key] bf16
    #pragma unroll
    for (int t = 0; t < 8; ++t) {
      #pragma unroll
      for (int r = 0; r < 8; ++r) {
        Ps[(wave * 16 + half * 8 + r) * QB + t * 16 + lm] = (bf16_t)s[t][r];
      }
    }

    // ---- O += P * V  (reads only this wave's P rows; wave-coherent LDS) ----
    const bf16_t* prow = Ps + (wave * 16 + lm) * QB;
    #pragma unroll
    for (int n = 0; n < 8; ++n) {
      const bf16_t* vrow = Vts + (n * 16 + lm) * QB;
      v8f acc = o_acc[n];
      #pragma unroll
      for (int c = 0; c < 4; ++c) {
        v16bf a  = ld_a(prow, c, half);
        v16bf bb = ld_b(vrow, c, half);
        acc = __builtin_amdgcn_wmma_f32_16x16x32_bf16(
            false, a, false, bb, (short)0, acc, false, false);
      }
      o_acc[n] = acc;
    }
  }

  // ---- normalize + store O: out[t][h*128 + d], t = qrow0 + local row ----
  #pragma unroll
  for (int r = 0; r < 8; ++r) {
    float inv = 1.0f / l_r[r];
    size_t t = qrow0 + (size_t)(wave * 16 + half * 8 + r);
    float* orow = out + t * (size_t)(H_ * D_) + h * D_;
    #pragma unroll
    for (int n = 0; n < 8; ++n) {
      orow[n * 16 + lm] = o_acc[n][r] * inv;
    }
  }
}

extern "C" void kernel_launch(void* const* d_in, const int* in_sizes, int n_in,
                              void* d_out, int out_size, void* d_ws, size_t ws_size,
                              hipStream_t stream) {
  const float* q     = (const float*)d_in[0];
  const float* k     = (const float*)d_in[1];
  const float* v     = (const float*)d_in[2];
  const int*   slots = (const int*)d_in[5];
  (void)in_sizes; (void)n_in; (void)d_ws; (void)ws_size; (void)out_size;

  float* out   = (float*)d_out;
  float* o_out = out;                                      // [8192, 2048]
  float* kc    = out + (size_t)(B_ * S_) * (H_ * D_);      // [16384, 4, 128]
  float* vc    = kc + (size_t)16384 * KV_ * D_;            // [16384, 4, 128]

  // zero both caches (contiguous region), then scatter tokens into them
  {
    int n4 = (2 * 16384 * KV_ * D_) / 4;  // 4,194,304 float4
    zero_kernel<<<(n4 + 255) / 256, 256, 0, stream>>>((float4*)kc, n4);
  }
  {
    int n4 = (B_ * S_) * (KV_ * D_ / 4);  // 1,048,576 float4
    scatter_kernel<<<(n4 + 255) / 256, 256, 0, stream>>>(
        (const float4*)k, (const float4*)v, slots, (float4*)kc, (float4*)vc);
  }

  // staircase flash attention: one workgroup per (b, h, q-block)
  size_t lds_bytes = (size_t)3 * 128 * 128 * sizeof(bf16_t);  // 96 KB
  attn_kernel<<<B_ * H_ * NQB, 256, lds_bytes, stream>>>(q, k, v, o_out);
}
